// CentroidDistance_5823975653995
// MI455X (gfx1250) — matrix operand order repaired and checked by reading, not compiled, not run on hardware
//
#include <hip/hip_runtime.h>
#include <math.h>

// CDNA5 / gfx1250 fused centroid-distance + segment-mean.
// dist = sqrt(||x||^2 + ||c||^2 - 2 x.c), x.c via v_wmma_f32_16x16x32_f16
// (f16 inputs, f32 accumulate; the exact fp32 norm terms dominate numerically).
// Centroids are pre-converted once to an fp16 copy in workspace (if it fits),
// removing ~2/3 of inner-loop VALU converts and halving B-side L2 traffic.
// Sorted `graph` lets each 32-row tile emit ~1 atomic per (graph,col).

typedef __attribute__((ext_vector_type(16))) _Float16 v16h;
typedef __attribute__((ext_vector_type(8)))  _Float16 v8h;
typedef __attribute__((ext_vector_type(4)))  _Float16 v4h;
typedef __attribute__((ext_vector_type(8)))  float    v8f;
typedef __attribute__((ext_vector_type(4)))  float    v4f;

#define KD 256   // feature dim (specialized fast path)
#define KC 512   // number of centroids (specialized fast path)

// ---------------- init: zero output + counts ----------------
__global__ __launch_bounds__(256) void cd_init(float* out, int* counts, int GC, int G) {
    int i = blockIdx.x * 256 + threadIdx.x;
    if (i < GC) out[i] = 0.0f;
    if (i < G)  counts[i] = 0;
}

// ---------------- per-centroid ||c||^2 in fp32 ----------------
__global__ __launch_bounds__(64) void cd_csq(const float* __restrict__ cw,
                                             float* __restrict__ csq, int D) {
    __shared__ float sm[64];
    int c = blockIdx.x, t = threadIdx.x;
    const float* p = cw + (size_t)c * D;
    float s = 0.f;
    for (int i = t; i < D; i += 64) { float v = p[i]; s += v * v; }
    sm[t] = s;
    __syncthreads();
    if (t == 0) {
        float tot = 0.f;
        for (int i = 0; i < 64; ++i) tot += sm[i];
        csq[c] = tot;
    }
}

// ---------------- one-time fp32 -> fp16 convert of centroids ----------------
__global__ __launch_bounds__(256) void cd_cvtB(const float* __restrict__ cw,
                                               _Float16* __restrict__ cwh, int n) {
    int i = (blockIdx.x * 256 + threadIdx.x) * 4;
    if (i + 3 < n) {
        v4f u = *(const v4f*)(cw + i);
        v4h h;
        #pragma unroll
        for (int j = 0; j < 4; ++j) h[j] = (_Float16)u[j];
        *(v4h*)(cwh + i) = h;
    } else {
        for (int j = i; j < n; ++j) cwh[j] = (_Float16)cw[j];
    }
}

// ---------------- per-graph node counts ----------------
__global__ __launch_bounds__(256) void cd_count(const int* __restrict__ graph,
                                                int* counts, int N) {
    int i = blockIdx.x * 256 + threadIdx.x;
    if (i < N) atomicAdd(&counts[graph[i]], 1);
}

// ---------------- main fused WMMA kernel ----------------
// block: 32 rows x 512 cols. 8 waves; wave w owns cols [64w, 64w+64),
// 2 row-tiles x 4 col-tiles of 16x16, K-loop over D in steps of 32.
// F16B: B fragments come from the pre-converted fp16 centroid copy.
template <bool F16B>
__global__ __launch_bounds__(256) void cd_main(const float*    __restrict__ x,
                                               const float*    __restrict__ cw,
                                               const _Float16* __restrict__ cwh,
                                               const int*      __restrict__ graph,
                                               const float*    __restrict__ csq,
                                               float*          __restrict__ out) {
    __shared__ float s_part[256];
    __shared__ float s_xsq[32];
    __shared__ int   s_g[32];

    const int tid     = threadIdx.x;
    const int rowbase = blockIdx.x * 32;

    // exact fp32 ||x||^2 for the 32 rows of this block
    {
        int r = tid >> 3, seg = tid & 7;
        const float* p = x + (size_t)(rowbase + r) * KD + seg * (KD / 8);
        float s = 0.f;
        #pragma unroll
        for (int i = 0; i < KD / 8; ++i) { float v = p[i]; s += v * v; }
        s_part[tid] = s;
    }
    if (tid < 32) s_g[tid] = graph[rowbase + tid];
    __syncthreads();
    if ((tid & 7) == 0) {
        int r = tid >> 3;
        float t = 0.f;
        #pragma unroll
        for (int i = 0; i < 8; ++i) t += s_part[r * 8 + i];
        s_xsq[r] = t;
    }
    __syncthreads();

    const int wave    = tid >> 5;   // wave32
    const int lane    = tid & 31;
    const int hi      = lane >> 4;  // lane half (selects K sub-ranges)
    const int lm      = lane & 15;  // M for A, N for B
    const int colbase = wave * 64;

    v8f acc[2][4] = {};

    const float* a0 = x + (size_t)(rowbase + lm) * KD;  // row tile 0
    const float* a1 = a0 + 16 * KD;                     // row tile 1

    #pragma unroll 2
    for (int k0 = 0; k0 < KD; k0 += 32) {
        // A fragments: 16-bit A 16x32 layout (contiguous, matching B):
        //   lanes<16: elems 0..15 -> K=k0..k0+15
        //   lanes>=16: elems 0..15 -> K=k0+16..k0+31.
        v16h A[2];
        {
            const v4f* p = (const v4f*)(a0 + k0 + hi * 16);
            v4f u0 = __builtin_nontemporal_load(p);
            v4f u1 = __builtin_nontemporal_load(p + 1);
            v4f u2 = __builtin_nontemporal_load(p + 2);
            v4f u3 = __builtin_nontemporal_load(p + 3);
            #pragma unroll
            for (int j = 0; j < 4; ++j) {
                A[0][j]      = (_Float16)u0[j];
                A[0][4 + j]  = (_Float16)u1[j];
                A[0][8 + j]  = (_Float16)u2[j];
                A[0][12 + j] = (_Float16)u3[j];
            }
            p = (const v4f*)(a1 + k0 + hi * 16);
            u0 = __builtin_nontemporal_load(p);
            u1 = __builtin_nontemporal_load(p + 1);
            u2 = __builtin_nontemporal_load(p + 2);
            u3 = __builtin_nontemporal_load(p + 3);
            #pragma unroll
            for (int j = 0; j < 4; ++j) {
                A[1][j]      = (_Float16)u0[j];
                A[1][4 + j]  = (_Float16)u1[j];
                A[1][8 + j]  = (_Float16)u2[j];
                A[1][12 + j] = (_Float16)u3[j];
            }
        }
        #pragma unroll
        for (int ct = 0; ct < 4; ++ct) {
            // B fragment: 16-bit B 32x16 layout: lane = N; lanes<16 hold
            // K=k0..k0+15 contiguous, lanes>=16 hold K=k0+16..k0+31.
            const size_t boff = (size_t)(colbase + ct * 16 + lm) * KD + k0 + hi * 16;
            v16h B;
            if constexpr (F16B) {
                const v8h* q = (const v8h*)(cwh + boff);
                v8h b0 = q[0], b1 = q[1];
                #pragma unroll
                for (int j = 0; j < 8; ++j) { B[j] = b0[j]; B[8 + j] = b1[j]; }
            } else {
                const v4f* q = (const v4f*)(cw + boff);
                v4f b0 = q[0], b1 = q[1], b2 = q[2], b3 = q[3];
                #pragma unroll
                for (int j = 0; j < 4; ++j) {
                    B[j]      = (_Float16)b0[j];
                    B[4 + j]  = (_Float16)b1[j];
                    B[8 + j]  = (_Float16)b2[j];
                    B[12 + j] = (_Float16)b3[j];
                }
            }
            acc[0][ct] = __builtin_amdgcn_wmma_f32_16x16x32_f16(
                false, A[0], false, B, (short)0, acc[0][ct], false, false);
            acc[1][ct] = __builtin_amdgcn_wmma_f32_16x16x32_f16(
                false, A[1], false, B, (short)0, acc[1][ct], false, false);
        }
    }

    // Epilogue: dist = sqrt(max(xsq + csq - 2 dot, 0)), segment-reduced.
    // C/D layout: elem i of lane -> row = rt*16 + hi*8 + i, col = base + lm.
    const bool uniform = (s_g[0] == s_g[31]);  // graph is sorted
    #pragma unroll
    for (int ct = 0; ct < 4; ++ct) {
        const int col = colbase + ct * 16 + lm;
        const float cs = csq[col];
        if (uniform) {
            float s = 0.f;
            #pragma unroll
            for (int rt = 0; rt < 2; ++rt)
                #pragma unroll
                for (int i = 0; i < 8; ++i) {
                    int row  = rt * 16 + hi * 8 + i;
                    float sq = s_xsq[row] + cs - 2.0f * acc[rt][ct][i];
                    s += sqrtf(fmaxf(sq, 0.0f));
                }
            s += __shfl_xor(s, 16, 32);          // combine the two lane halves
            if (hi == 0)
                atomicAdd(&out[(size_t)s_g[0] * KC + col], s);
        } else {
            // rare tile spanning >1 graph: run-length compacted atomics
            int   curg = s_g[hi * 8];
            float run  = 0.f;
            #pragma unroll
            for (int rt = 0; rt < 2; ++rt)
                #pragma unroll
                for (int i = 0; i < 8; ++i) {
                    int row  = rt * 16 + hi * 8 + i;
                    int gg   = s_g[row];
                    float sq = s_xsq[row] + cs - 2.0f * acc[rt][ct][i];
                    float dv = sqrtf(fmaxf(sq, 0.0f));
                    if (gg != curg) {
                        atomicAdd(&out[(size_t)curg * KC + col], run);
                        run = 0.f; curg = gg;
                    }
                    run += dv;
                }
            atomicAdd(&out[(size_t)curg * KC + col], run);
        }
    }
}

// ---------------- generic scalar fallback / tail rows ----------------
__global__ __launch_bounds__(256) void cd_fallback(const float* __restrict__ x,
                                                   const float* __restrict__ cw,
                                                   const int*   __restrict__ graph,
                                                   const float* __restrict__ csq,
                                                   float* out, int row0, int N,
                                                   int D, int C) {
    int row = row0 + blockIdx.x;
    if (row >= N) return;
    int g = graph[row];
    const float* xp = x + (size_t)row * D;
    for (int col = threadIdx.x; col < C; col += blockDim.x) {
        const float* cp = cw + (size_t)col * D;
        float dot = 0.f, xs = 0.f;
        for (int k = 0; k < D; ++k) {
            float xv = xp[k], cv = cp[k];
            dot += xv * cv; xs += xv * xv;
        }
        float d = sqrtf(fmaxf(xs + csq[col] - 2.0f * dot, 0.0f));
        atomicAdd(&out[(size_t)g * C + col], d);
    }
}

// ---------------- finalize: divide by counts ----------------
__global__ __launch_bounds__(256) void cd_fin(float* out, const int* counts,
                                              int GC, int C) {
    int i = blockIdx.x * 256 + threadIdx.x;
    if (i < GC) {
        int c = counts[i / C];
        out[i] = out[i] / (float)(c > 0 ? c : 1);
    }
}

extern "C" void kernel_launch(void* const* d_in, const int* in_sizes, int n_in,
                              void* d_out, int out_size, void* d_ws, size_t ws_size,
                              hipStream_t stream) {
    const float* x     = (const float*)d_in[0];
    const float* cw    = (const float*)d_in[1];
    const int*   graph = (const int*)d_in[2];

    int N = in_sizes[2];
    int D = (N > 0) ? in_sizes[0] / N : KD;
    int C = (D > 0) ? in_sizes[1] / D : KC;
    int GC = out_size;
    int G = (C > 0) ? GC / C : 0;

    float* out = (float*)d_out;

    // workspace layout: csq (C f32) | counts (G i32) | cwh (C*D f16, 16B aligned)
    float* csq    = (float*)d_ws;
    int*   counts = (int*)((char*)d_ws + (size_t)C * sizeof(float));
    size_t base   = ((size_t)C * sizeof(float) + (size_t)G * sizeof(int) + 15) & ~(size_t)15;
    _Float16* cwh = (_Float16*)((char*)d_ws + base);
    size_t need_f16 = base + (size_t)C * D * sizeof(_Float16);
    bool use_f16b = (ws_size >= need_f16);

    cd_init<<<(GC + 255) / 256, 256, 0, stream>>>(out, counts, GC, G);
    cd_csq<<<C, 64, 0, stream>>>(cw, csq, D);
    cd_count<<<(N + 255) / 256, 256, 0, stream>>>(graph, counts, N);
    if (use_f16b) {
        int n = C * D;
        cd_cvtB<<<(n / 4 + 255) / 256 + 1, 256, 0, stream>>>(cw, cwh, n);
    }

    if (D == KD && C == KC) {
        int nfull = N / 32;
        if (nfull > 0) {
            if (use_f16b)
                cd_main<true><<<nfull, 256, 0, stream>>>(x, cw, cwh, graph, csq, out);
            else
                cd_main<false><<<nfull, 256, 0, stream>>>(x, cw, cwh, graph, csq, out);
        }
        int row0 = nfull * 32, rem = N - row0;
        if (rem > 0)
            cd_fallback<<<rem, 256, 0, stream>>>(x, cw, graph, csq, out, row0, N, D, C);
    } else {
        cd_fallback<<<N, 256, 0, stream>>>(x, cw, graph, csq, out, 0, N, D, C);
    }

    cd_fin<<<(GC + 255) / 256, 256, 0, stream>>>(out, counts, GC, C);
}
